// CharAttention_86715389706665
// MI455X (gfx1250) — matrix operand order repaired and checked by reading, not compiled, not run on hardware
//
#include <hip/hip_runtime.h>
#include <hip/hip_bf16.h>

// ---------------------------------------------------------------------------
// CDNA5 (gfx1250) implementation of windowed char-attention.
// Only o[last_ix] survives the gather => Q is one row per window; heavy math
// (K/V projection 137 GFLOP, final projection 2 GFLOP) runs on
// v_wmma_f32_16x16x32_bf16 with 2x2 register blocking for operand reuse.
// ---------------------------------------------------------------------------

typedef __attribute__((ext_vector_type(16))) __bf16        v16bf;
typedef __attribute__((ext_vector_type(8)))  float         v8f;
typedef __attribute__((ext_vector_type(4)))  unsigned int  u32x4;

union V16BF { u32x4 u[2]; v16bf v; };   // two 16B chunks -> one bf16 A/B fragment

#define C_DIM   1024
#define SEQ     32
#define HEADS   16
#define HD      64
#define BW      1024          // B*W
#define KV_COLS 2048          // [K | V]

static __device__ __forceinline__ v16bf load_frag(const __bf16* p0, const __bf16* p1) {
    V16BF f;
    f.u[0] = *(const u32x4*)p0;
    f.u[1] = *(const u32x4*)p1;
    return f.v;
}

// ---------------------------------------------------------------------------
// Kernel 0: transpose + convert weights fp32 -> bf16.
//   wT[f][e] = attn_w[e][f]  (rows 0..1023 = Wq^T, 1024.. = Wk^T, 2048.. = Wv^T)
//   pT[f][e] = proj_w[e][f]
// Row-per-output-column makes every WMMA B-fragment two contiguous 16B lane loads.
// ---------------------------------------------------------------------------
__global__ __launch_bounds__(256) void wconv_kernel(const float* __restrict__ attn_w,
                                                    const float* __restrict__ proj_w,
                                                    __bf16* __restrict__ wT,
                                                    __bf16* __restrict__ pT)
{
    long idx = (long)blockIdx.x * 256 + threadIdx.x;
    const long NA = 3072l * 1024;
    const long NP = 1024l * 1024;
    if (idx < NA) {
        long f = idx >> 10, e = idx & 1023;
        wT[idx] = (__bf16)attn_w[e * 3072 + f];
    } else if (idx < NA + NP) {
        long i = idx - NA;
        long f = i >> 10, e = i & 1023;
        pT[i] = (__bf16)proj_w[e * 1024 + f];
    }
}

// ---------------------------------------------------------------------------
// Kernel 1: one workgroup (256 thr = 8 waves) per (b,w) window.
// ---------------------------------------------------------------------------
__global__ __launch_bounds__(256) void window_attn_kernel(
    const float*  __restrict__ x,        // [BW][32][1024]
    const int*    __restrict__ mask,     // [BW][32]
    const float*  __restrict__ pos_emb,  // [256][1024]
    const __bf16* __restrict__ wT,       // [3072][1024] transposed bf16 weights
    __bf16*       __restrict__ G)        // [BW][1024]
{
    extern __shared__ char smem[];
    __bf16* xl  = (__bf16*)smem;                                        // 32*1024 bf16
    __bf16* kv  = (__bf16*)(smem + SEQ * C_DIM * 2);                    // 32*2048 bf16
    float*  qv  = (float*)(smem + SEQ * C_DIM * 2 + SEQ * KV_COLS * 2); // 1024 f32
    float*  sc  = qv + C_DIM;                                           // 16*32 f32
    int*    lix = (int*)(sc + HEADS * SEQ);

    const int tid  = threadIdx.x;
    const int lane = tid & 31;
    const int wave = tid >> 5;
    const int bw   = blockIdx.x;
    const int w    = bw & 255;

    if (tid == 0) {
        int s = 0;
        const int* m = mask + bw * SEQ;
        for (int j = 0; j < SEQ; ++j) s += m[j];
        *lix = s - 1;
    }
    // ---- x tile -> bf16 LDS (coalesced) ----
    const float* xp = x + (long)bw * SEQ * C_DIM;
    for (int i = tid; i < SEQ * C_DIM; i += 256) xl[i] = (__bf16)xp[i];
    __syncthreads();

    const int li = *lix;

    // ---- q = x[li] @ Wq  (one row; 4 output dims per thread) ----
    {
        const __bf16* xrow = xl + li * C_DIM;
        for (int r = 0; r < 4; ++r) {
            int f = tid * 4 + r;
            const __bf16* wrow = wT + (long)f * C_DIM;
            float acc = 0.f;
            for (int e = 0; e < C_DIM; ++e) acc += (float)xrow[e] * (float)wrow[e];
            qv[f] = acc;
        }
    }

    // ---- K|V = x @ [Wk|Wv] via WMMA bf16 (M=32, N=2048, K=1024) ----
    // 2 M-tiles x 2 N-tiles per wave iteration: 4 WMMAs per k-step from
    // 2 A-frag + 2 B-frag loads (max operand reuse per MAC).
    const int laneM  = lane & 15;
    const int kbHalf = (lane >> 4) << 3;    // A: 0 or 8
    const int kbB    = (lane >> 4) << 4;    // B: 0 or 16
    for (int np = wave; np < (KV_COLS / 16) / 2; np += 8) {
        const int nt0 = np * 2;
        v8f a00 = {}, a01 = {}, a10 = {}, a11 = {};
        const __bf16* brow0 = wT + (long)(1024 + nt0 * 16 + laneM) * C_DIM + kbHalf;
        const __bf16* brow1 = brow0 + 16 * C_DIM;
        for (int ks = 0; ks < C_DIM / 32; ++ks) {
            // prefetch weight stream a few k-steps ahead (global_prefetch_b8)
            __builtin_prefetch(brow0 + (ks + 4) * 32, 0, 1);
            __builtin_prefetch(brow1 + (ks + 4) * 32, 0, 1);

            v16bf bf0 = load_frag(brow0 + ks * 32, brow0 + ks * 32 + 16);
            v16bf bf1 = load_frag(brow1 + ks * 32, brow1 + ks * 32 + 16);

            // A fragments (ISA 16-bit A layout): lane M = lane%16,
            // chunks at K = kb..kb+7 and kb+16..kb+23, kb = (lane<16 ? 0 : 8)
            const __bf16* a0p = xl + laneM * C_DIM + ks * 32 + kbHalf;
            const __bf16* a1p = a0p + 16 * C_DIM;
            v16bf af0 = load_frag(a0p, a0p + 16);
            v16bf af1 = load_frag(a1p, a1p + 16);

            a00 = __builtin_amdgcn_wmma_f32_16x16x32_bf16(false, af0, false, bf0,
                                                          (short)0, a00, false, false);
            a10 = __builtin_amdgcn_wmma_f32_16x16x32_bf16(false, af1, false, bf0,
                                                          (short)0, a10, false, false);
            a01 = __builtin_amdgcn_wmma_f32_16x16x32_bf16(false, af0, false, bf1,
                                                          (short)0, a01, false, false);
            a11 = __builtin_amdgcn_wmma_f32_16x16x32_bf16(false, af1, false, bf1,
                                                          (short)0, a11, false, false);
        }
        // D layout: VGPR r -> M = r (lanes 0-15) / r+8 (lanes 16-31), N = lane%16
        int col0  = nt0 * 16 + laneM;
        int rbase = kbHalf;
        for (int r = 0; r < 8; ++r) {
            kv[(r + rbase) * KV_COLS + col0]           = (__bf16)a00[r];
            kv[(16 + r + rbase) * KV_COLS + col0]      = (__bf16)a10[r];
            kv[(r + rbase) * KV_COLS + col0 + 16]      = (__bf16)a01[r];
            kv[(16 + r + rbase) * KV_COLS + col0 + 16] = (__bf16)a11[r];
        }
    }
    __syncthreads();

    // ---- scores for the single q row (causal: keys j <= li) ----
    const float scale = 0.125f;   // 1/sqrt(64)
    for (int s = tid; s < HEADS * SEQ; s += 256) {
        int h = s >> 5, j = s & 31;
        const __bf16* krow = kv + j * KV_COLS + h * HD;
        const float*  qh   = qv + h * HD;
        float acc = 0.f;
        for (int d = 0; d < HD; ++d) acc += qh[d] * (float)krow[d];
        sc[s] = (j <= li) ? acc * scale : -3.0e38f;
    }
    __syncthreads();

    // ---- per-head softmax (16 heads, one thread each) ----
    if (tid < HEADS) {
        float* row = sc + tid * SEQ;
        float mx = -3.0e38f;
        for (int j = 0; j < SEQ; ++j) mx = fmaxf(mx, row[j]);
        float sum = 0.f;
        for (int j = 0; j < SEQ; ++j) { float e = __expf(row[j] - mx); row[j] = e; sum += e; }
        float inv = 1.f / sum;
        for (int j = 0; j < SEQ; ++j) row[j] *= inv;
    }
    __syncthreads();

    // ---- o = attn @ V (one row), add pos_emb, store gathered row bf16 ----
    for (int r = 0; r < 4; ++r) {
        int f = tid * 4 + r;
        int h = f >> 6, d = f & 63;
        const float* a = sc + h * SEQ;
        float acc = 0.f;
        for (int j = 0; j < SEQ; ++j)
            acc += a[j] * (float)kv[j * KV_COLS + 1024 + h * HD + d];
        G[(long)bw * C_DIM + f] = (__bf16)(acc + pos_emb[(long)w * C_DIM + f]);
    }
}

// ---------------------------------------------------------------------------
// Kernel 2: out[1024][1024] = G (bf16) @ proj_w via proj^T (bf16).
// One wave per 32x32 output super-tile (2x2 WMMA tiles), 8 waves per block.
// ---------------------------------------------------------------------------
__global__ __launch_bounds__(256) void proj_kernel(
    const __bf16* __restrict__ G,    // [1024][1024]
    const __bf16* __restrict__ pT,   // [1024][1024] proj^T
    float*        __restrict__ out)  // [1024][1024]
{
    const int lane  = threadIdx.x & 31;
    const int wave  = threadIdx.x >> 5;
    const int st    = blockIdx.x * 8 + wave;      // 1024 super-tiles (32x32)
    const int smt   = st >> 5, snt = st & 31;

    const int laneM  = lane & 15;
    const int kbHalf = (lane >> 4) << 3;
    const int kbB    = (lane >> 4) << 4;

    const __bf16* arow0 = G  + (long)(smt * 32 + laneM) * C_DIM;
    const __bf16* arow1 = arow0 + 16 * C_DIM;
    const __bf16* brow0 = pT + (long)(snt * 32 + laneM) * C_DIM + kbHalf;
    const __bf16* brow1 = brow0 + 16 * C_DIM;

    v8f a00 = {}, a01 = {}, a10 = {}, a11 = {};
    for (int ks = 0; ks < C_DIM / 32; ++ks) {
        __builtin_prefetch(brow0 + (ks + 4) * 32, 0, 1);
        __builtin_prefetch(arow0 + (ks + 4) * 32, 0, 1);

        int kb = ks * 32 + kbHalf;
        v16bf af0 = load_frag(arow0 + kb, arow0 + kb + 16);
        v16bf af1 = load_frag(arow1 + kb, arow1 + kb + 16);
        v16bf bf0 = load_frag(brow0 + ks * 32, brow0 + ks * 32 + 16);
        v16bf bf1 = load_frag(brow1 + ks * 32, brow1 + ks * 32 + 16);

        a00 = __builtin_amdgcn_wmma_f32_16x16x32_bf16(false, af0, false, bf0,
                                                      (short)0, a00, false, false);
        a10 = __builtin_amdgcn_wmma_f32_16x16x32_bf16(false, af1, false, bf0,
                                                      (short)0, a10, false, false);
        a01 = __builtin_amdgcn_wmma_f32_16x16x32_bf16(false, af0, false, bf1,
                                                      (short)0, a01, false, false);
        a11 = __builtin_amdgcn_wmma_f32_16x16x32_bf16(false, af1, false, bf1,
                                                      (short)0, a11, false, false);
    }
    // D layout: VGPR r -> M = r (lanes 0-15) / r+8 (lanes 16-31), N = lane%16
    int row = smt * 32 + kbHalf;
    int col = snt * 32 + laneM;
    for (int r = 0; r < 8; ++r) {
        out[(long)(row + r) * C_DIM + col]           = a00[r];
        out[(long)(row + 16 + r) * C_DIM + col]      = a10[r];
        out[(long)(row + r) * C_DIM + col + 16]      = a01[r];
        out[(long)(row + 16 + r) * C_DIM + col + 16] = a11[r];
    }
}

// ---------------------------------------------------------------------------
extern "C" void kernel_launch(void* const* d_in, const int* in_sizes, int n_in,
                              void* d_out, int out_size, void* d_ws, size_t ws_size,
                              hipStream_t stream) {
    const float* x      = (const float*)d_in[0];   // [4][256][32][1024]
    const int*   mask   = (const int*)  d_in[1];   // [4][256][32]
    const float* pos    = (const float*)d_in[2];   // [256][1024]
    const float* attn_w = (const float*)d_in[3];   // [1024][3072]
    const float* proj_w = (const float*)d_in[4];   // [1024][1024]
    float*       out    = (float*)d_out;           // [1024][1024]

    // workspace layout (bf16): wT 6MB | pT 2MB | G 2MB
    __bf16* wT = (__bf16*)d_ws;
    __bf16* pT = wT + 3072l * 1024;
    __bf16* G  = pT + 1024l * 1024;

    // K0: weight transpose+convert (4M elements / 256 threads)
    wconv_kernel<<<(3072 * 1024 + 1024 * 1024) / 256, 256, 0, stream>>>(attn_w, proj_w, wT, pT);

    // K1: per-window attention. Dynamic LDS: x(64K) + kv(128K) + q(4K) + sc(2K) + flags
    size_t smem = (size_t)SEQ * C_DIM * 2 + (size_t)SEQ * KV_COLS * 2
                + C_DIM * 4 + HEADS * SEQ * 4 + 16;
    window_attn_kernel<<<BW, 256, smem, stream>>>(x, mask, pos, wT, G);

    // K2: final projection GEMM (1024 super-tiles / 8 waves per block)
    proj_kernel<<<128, 256, 0, stream>>>(G, pT, out);
}
